// GRUEncoderAtt_52381421142200
// MI455X (gfx1250) — matrix operand order repaired and checked by reading, not compiled, not run on hardware
//
#include <hip/hip_runtime.h>
#include <hip/hip_bf16.h>

// ---------------------------------------------------------------------------
// GRU encoder with visual attention, bf16-WMMA GEMMs on gfx1250 (MI455X).
// B=512, T=32, D_TOK=512, V_NUM=36, V_DIM=2048, V_COMP=512, H=1024, D_IN=1024
// GEMMs: LDS-staged via GLOBAL_LOAD_ASYNC_TO_LDS (ASYNCcnt), double-buffered,
// 128x128 block tile, 8 waves, 32x64 wave tile, 8 WMMAs per K=32 step.
// ---------------------------------------------------------------------------

#define B_   512
#define T_   32
#define DTOK 512
#define VN   36
#define VD   2048
#define VC   512
#define H_   1024
#define DIN  1024
#define G3   3072   // 3*H

#define BM   128
#define BN   128
#define BK   32
#define LSTR 40     // padded LDS row stride in elements (80 B, 16B-aligned)

typedef __attribute__((ext_vector_type(16))) __bf16 v16bf;
typedef __attribute__((ext_vector_type(8)))  __bf16 v8bf;
typedef __attribute__((ext_vector_type(8)))  float  v8f;

__device__ __forceinline__ unsigned short f32_to_bf16(float f) {
  unsigned int u = __float_as_uint(f);
  u += 0x7FFFu + ((u >> 16) & 1u);   // round-to-nearest-even
  return (unsigned short)(u >> 16);
}

// ---- CDNA5 async global->LDS copy (ASYNCcnt-tracked), via inline asm ------
__device__ __forceinline__ void async_b128(unsigned lds_off,
                                           const unsigned short* gptr) {
  asm volatile("global_load_async_to_lds_b128 %0, %1, off"
               :: "v"(lds_off), "v"(gptr) : "memory");
}
__device__ __forceinline__ void wait_async0() {
  asm volatile("s_wait_asynccnt 0x0" ::: "memory");
}

// ------------------------- one-time weight prep ----------------------------

__global__ void k_cvt_bf16(const float* __restrict__ src,
                           unsigned short* __restrict__ dst, int n) {
  int i = blockIdx.x * blockDim.x + threadIdx.x;
  if (i < n) dst[i] = f32_to_bf16(src[i]);
}

// dst[n*K + k] = bf16(src[k*N + n])   (Wq is stored [K,N]; we want [N,K])
__global__ void k_cvt_bf16_T(const float* __restrict__ src,
                             unsigned short* __restrict__ dst, int K, int N) {
  int i = blockIdx.x * blockDim.x + threadIdx.x;
  if (i < K * N) {
    int n = i / K, k = i - n * K;
    dst[i] = f32_to_bf16(src[(size_t)k * N + n]);
  }
}

__global__ void k_zero_h(float* __restrict__ hf,
                         unsigned short* __restrict__ hb, int n) {
  int i = blockIdx.x * blockDim.x + threadIdx.x;
  if (i < n) { hf[i] = 0.0f; hb[i] = 0; }
}

// ------------------------------ WMMA GEMM ----------------------------------
// C[M,N] (f32) = A[M,K] (bf16 row-major) * W[N,K]^T (bf16 row-major).
// Block 256 threads = 8 waves (4 along M x 2 along N); wave tile 32x64.
// K tiles staged in LDS by the async engine; one barrier per K-step.
__global__ __launch_bounds__(256)
void k_gemm_bf16_wmma(const unsigned short* __restrict__ A,
                      const unsigned short* __restrict__ W,
                      float* __restrict__ C,
                      int M, int N, int K) {
  __shared__ __align__(16) unsigned short As[2][BM * LSTR];
  __shared__ __align__(16) unsigned short Bs[2][BN * LSTR];

  const int tid  = threadIdx.x;
  const int lane = tid & 31;
  const int wave = tid >> 5;
  const int wm   = wave & 3;            // 4 waves along M (32 rows each)
  const int wn   = wave >> 2;           // 2 waves along N (64 cols each)
  const int mBlk = blockIdx.y * BM;
  const int nBlk = blockIdx.x * BN;
  const int lr   = lane & 15;
  const int lhi  = lane >> 4;           // K-half selector

  // async-copy mapping: 512 16B chunks per tile, 2 per thread per operand
  const int r0 = tid >> 2, c0 = tid & 3;          // chunk (row, col16B)
  const int r1 = (tid + 256) >> 2, c1 = (tid + 256) & 3;

  auto issue_tile = [&](int kb, int buf) {
    const int kOff = kb * BK;
    async_b128((unsigned)(size_t)&As[buf][r0 * LSTR + c0 * 8],
               A + (size_t)(mBlk + r0) * K + kOff + c0 * 8);
    async_b128((unsigned)(size_t)&As[buf][r1 * LSTR + c1 * 8],
               A + (size_t)(mBlk + r1) * K + kOff + c1 * 8);
    async_b128((unsigned)(size_t)&Bs[buf][r0 * LSTR + c0 * 8],
               W + (size_t)(nBlk + r0) * K + kOff + c0 * 8);
    async_b128((unsigned)(size_t)&Bs[buf][r1 * LSTR + c1 * 8],
               W + (size_t)(nBlk + r1) * K + kOff + c1 * 8);
  };

  v8f acc[2][4];
#pragma unroll
  for (int mt = 0; mt < 2; ++mt)
#pragma unroll
    for (int nt = 0; nt < 4; ++nt) acc[mt][nt] = (v8f){};

  const int nk = K / BK;
  issue_tile(0, 0);

  for (int kb = 0; kb < nk; ++kb) {
    wait_async0();          // my async writes (tile kb) done
    __syncthreads();        // everyone's writes done; prev ds reads retired
    if (kb + 1 < nk) issue_tile(kb + 1, (kb + 1) & 1);

    const unsigned short* as = As[kb & 1];
    const unsigned short* bs = Bs[kb & 1];

    union { v16bf v; v8bf h[2]; } a[2], b[4];
#pragma unroll
    for (int mt = 0; mt < 2; ++mt) {
      const int row = wm * 32 + mt * 16 + lr;
      a[mt].h[0] = *(const v8bf*)&as[row * LSTR + lhi * 8];
      a[mt].h[1] = *(const v8bf*)&as[row * LSTR + lhi * 8 + 16];
    }
#pragma unroll
    for (int nt = 0; nt < 4; ++nt) {
      const int row = wn * 64 + nt * 16 + lr;
      b[nt].h[0] = *(const v8bf*)&bs[row * LSTR + lhi * 8];
      b[nt].h[1] = *(const v8bf*)&bs[row * LSTR + lhi * 8 + 16];
    }
#pragma unroll
    for (int mt = 0; mt < 2; ++mt)
#pragma unroll
      for (int nt = 0; nt < 4; ++nt)
        acc[mt][nt] = __builtin_amdgcn_wmma_f32_16x16x32_bf16(
            false, a[mt].v, false, b[nt].v, (short)0, acc[mt][nt], false, false);
  }

  // C/D layout: VGPR r, lanes 0-15 -> (M=r, N=lane); lanes 16-31 -> (M=8+r).
#pragma unroll
  for (int mt = 0; mt < 2; ++mt) {
    const int mbase = mBlk + wm * 32 + mt * 16 + lhi * 8;
#pragma unroll
    for (int nt = 0; nt < 4; ++nt) {
      const int nbase = nBlk + wn * 64 + nt * 16 + lr;
#pragma unroll
      for (int r = 0; r < 8; ++r)
        C[(size_t)(mbase + r) * N + nbase] = acc[mt][nt][r];
    }
  }
}

// ----------------------------- attention -----------------------------------
// One block (256 threads) per batch element.  scores_n = <vf[b,n,:], q[b,:]>,
// softmax over 36 regions, v = sum_n p_n * vf[b,n,:], emitted as bf16.
__global__ __launch_bounds__(256)
void k_attention(const float* __restrict__ q,        // [B, VD]
                 const float* __restrict__ vf,       // [B, VN, VD]
                 unsigned short* __restrict__ vout)  // [B, VD] bf16
{
  const int b   = blockIdx.x;
  const int tid = threadIdx.x;
  const int c0  = tid * 8;                       // 8 columns per thread
  const float* qb = q  + (size_t)b * VD;
  const float* vb = vf + (size_t)b * VN * VD;

  float qv[8];
#pragma unroll
  for (int j = 0; j < 8; ++j) qv[j] = qb[c0 + j];

  float part[VN];
#pragma unroll
  for (int n = 0; n < VN; ++n) {
    const float* row = vb + (size_t)n * VD + c0;
    float s = 0.f;
#pragma unroll
    for (int j = 0; j < 8; ++j) s += row[j] * qv[j];
    part[n] = s;
  }
  // wave32 reduction
#pragma unroll
  for (int n = 0; n < VN; ++n) {
    float s = part[n];
    for (int off = 16; off > 0; off >>= 1) s += __shfl_down(s, off, 32);
    part[n] = s;
  }
  __shared__ float sred[8][VN];
  __shared__ float sscore[VN];
  const int wv = tid >> 5, ln = tid & 31;
  if (ln == 0) {
#pragma unroll
    for (int n = 0; n < VN; ++n) sred[wv][n] = part[n];
  }
  __syncthreads();
  if (tid < VN) {
    float s = 0.f;
#pragma unroll
    for (int w = 0; w < 8; ++w) s += sred[w][tid];
    sscore[tid] = s;
  }
  __syncthreads();

  float sc[VN], mx = -3.4e38f;
#pragma unroll
  for (int n = 0; n < VN; ++n) { sc[n] = sscore[n]; mx = fmaxf(mx, sc[n]); }
  float denom = 0.f;
#pragma unroll
  for (int n = 0; n < VN; ++n) { sc[n] = __expf(sc[n] - mx); denom += sc[n]; }
  const float inv = 1.0f / denom;

  float acc[8] = {0.f,0.f,0.f,0.f,0.f,0.f,0.f,0.f};
#pragma unroll
  for (int n = 0; n < VN; ++n) {
    const float p = sc[n] * inv;
    const float* row = vb + (size_t)n * VD + c0;
#pragma unroll
    for (int j = 0; j < 8; ++j) acc[j] += p * row[j];
  }
#pragma unroll
  for (int j = 0; j < 8; ++j)
    vout[(size_t)b * VD + c0 + j] = f32_to_bf16(acc[j]);
}

// --------------------- x = concat(tok_t, vc + b_vc) ------------------------
__global__ __launch_bounds__(256)
void k_build_x(const float* __restrict__ input_seq,  // [B, T, DTOK]
               const float* __restrict__ vc,         // [B, VC]
               const float* __restrict__ b_vc,       // [VC]
               unsigned short* __restrict__ x,       // [B, DIN] bf16
               int t) {
  const int i = blockIdx.x * blockDim.x + threadIdx.x;  // B*DIN
  const int b = i >> 10, j = i & 1023;
  float val;
  if (j < DTOK) val = input_seq[((size_t)b * T_ + t) * DTOK + j];
  else          val = vc[(size_t)b * VC + (j - DTOK)] + b_vc[j - DTOK];
  x[i] = f32_to_bf16(val);
}

// ------------------------------ GRU gates ----------------------------------
__global__ __launch_bounds__(256)
void k_gru_gate(const float* __restrict__ gi,       // [B, 3H]
                const float* __restrict__ gh,       // [B, 3H]
                const float* __restrict__ b_ih,     // [3H]
                const float* __restrict__ b_hh,     // [3H]
                const int*   __restrict__ lengths,  // [B]
                float* __restrict__ h_f32,          // [B, H] in/out
                unsigned short* __restrict__ h_bf16,// [B, H]
                float* __restrict__ outs,           // [B, T, H]
                float* __restrict__ h_last,         // [B, H]
                int t) {
  const int i = blockIdx.x * blockDim.x + threadIdx.x;  // B*H
  const int b = i >> 10, j = i & 1023;
  const size_t base = (size_t)b * G3;
  const float ir = gi[base + j]          + b_ih[j];
  const float iz = gi[base + H_ + j]     + b_ih[H_ + j];
  const float in_= gi[base + 2*H_ + j]   + b_ih[2*H_ + j];
  const float hr = gh[base + j]          + b_hh[j];
  const float hz = gh[base + H_ + j]     + b_hh[H_ + j];
  const float hn = gh[base + 2*H_ + j]   + b_hh[2*H_ + j];
  const float r = 1.f / (1.f + __expf(-(ir + hr)));
  const float z = 1.f / (1.f + __expf(-(iz + hz)));
  const float n = tanhf(in_ + r * hn);
  float hnew = (1.f - z) * n + z * h_f32[i];
  if (lengths[b] <= t) hnew = 0.f;
  h_f32[i]  = hnew;
  h_bf16[i] = f32_to_bf16(hnew);
  outs[((size_t)b * T_ + t) * H_ + j] = hnew;
  h_last[i] = hnew;
}

// ------------------------------- launcher ----------------------------------

extern "C" void kernel_launch(void* const* d_in, const int* in_sizes, int n_in,
                              void* d_out, int out_size, void* d_ws, size_t ws_size,
                              hipStream_t stream) {
  (void)in_sizes; (void)n_in; (void)out_size; (void)ws_size;

  const float* input_seq = (const float*)d_in[0];
  const int*   lengths   = (const int*)  d_in[1];
  const float* v_feature = (const float*)d_in[2];
  const float* Wq        = (const float*)d_in[3];   // [H, VD]  (== [K,N])
  const float* Wvc       = (const float*)d_in[4];   // [VC, VD] (== [N,K])
  const float* bvc       = (const float*)d_in[5];
  const float* Wih       = (const float*)d_in[6];   // [3H, DIN]
  const float* Whh       = (const float*)d_in[7];   // [3H, H]
  const float* bih       = (const float*)d_in[8];
  const float* bhh       = (const float*)d_in[9];

  float* outs  = (float*)d_out;                       // [B, T, H]
  float* hlast = (float*)d_out + (size_t)B_ * T_ * H_;// [B, H]

  // ---- workspace carve (256B aligned) ----
  char* ws = (char*)d_ws;
  auto carve = [&](size_t bytes) -> char* {
    char* p = ws; ws += (bytes + 255) & ~(size_t)255; return p;
  };
  unsigned short* WqT_b = (unsigned short*)carve((size_t)VD * H_ * 2); // [VD, H]
  unsigned short* Wvc_b = (unsigned short*)carve((size_t)VC * VD * 2);
  unsigned short* Wih_b = (unsigned short*)carve((size_t)G3 * DIN * 2);
  unsigned short* Whh_b = (unsigned short*)carve((size_t)G3 * H_ * 2);
  float*          h_f   = (float*)         carve((size_t)B_ * H_ * 4);
  unsigned short* h_b   = (unsigned short*)carve((size_t)B_ * H_ * 2);
  float*          q_f   = (float*)         carve((size_t)B_ * VD * 4);
  unsigned short* v_b   = (unsigned short*)carve((size_t)B_ * VD * 2);
  float*          vc_f  = (float*)         carve((size_t)B_ * VC * 4);
  unsigned short* x_b   = (unsigned short*)carve((size_t)B_ * DIN * 2);
  float*          gi_f  = (float*)         carve((size_t)B_ * G3 * 4);
  float*          gh_f  = (float*)         carve((size_t)B_ * G3 * 4);

  // ---- one-time weight conversion ----
  {
    int n;
    n = VD * H_;  k_cvt_bf16_T<<<(n + 255) / 256, 256, 0, stream>>>(Wq, WqT_b, H_, VD);
    n = VC * VD;  k_cvt_bf16  <<<(n + 255) / 256, 256, 0, stream>>>(Wvc, Wvc_b, n);
    n = G3 * DIN; k_cvt_bf16  <<<(n + 255) / 256, 256, 0, stream>>>(Wih, Wih_b, n);
    n = G3 * H_;  k_cvt_bf16  <<<(n + 255) / 256, 256, 0, stream>>>(Whh, Whh_b, n);
    n = B_ * H_;  k_zero_h    <<<(n + 255) / 256, 256, 0, stream>>>(h_f, h_b, n);
  }

  const dim3 blk(256);
  const dim3 grid_q (VD / BN, B_ / BM);   // (16, 4)
  const dim3 grid_vc(VC / BN, B_ / BM);   // ( 4, 4)
  const dim3 grid_g (G3 / BN, B_ / BM);   // (24, 4)
  const int  nBH = B_ * H_;
  const int  nBX = B_ * DIN;

  for (int t = 0; t < T_; ++t) {
    // q = h @ Wq                 [B,H] x [H,VD] -> [B,VD]
    k_gemm_bf16_wmma<<<grid_q, blk, 0, stream>>>(h_b, WqT_b, q_f, B_, VD, H_);
    // attention -> v (bf16)
    k_attention<<<B_, blk, 0, stream>>>(q_f, v_feature, v_b);
    // vc = v @ W_vc^T            [B,VD] x [VD,VC] -> [B,VC]
    k_gemm_bf16_wmma<<<grid_vc, blk, 0, stream>>>(v_b, Wvc_b, vc_f, B_, VC, VD);
    // x = concat(tok_t, vc + b_vc) -> bf16
    k_build_x<<<(nBX + 255) / 256, blk, 0, stream>>>(input_seq, vc_f, bvc, x_b, t);
    // gi = x @ W_ih^T ; gh = h @ W_hh^T
    k_gemm_bf16_wmma<<<grid_g, blk, 0, stream>>>(x_b, Wih_b, gi_f, B_, G3, DIN);
    k_gemm_bf16_wmma<<<grid_g, blk, 0, stream>>>(h_b, Whh_b, gh_f, B_, G3, H_);
    // gates + mask + write outs / h_last
    k_gru_gate<<<(nBH + 255) / 256, blk, 0, stream>>>(gi_f, gh_f, bih, bhh,
                                                      lengths, h_f, h_b,
                                                      outs, hlast, t);
  }
}
